// FunkNN_82566451299275
// MI455X (gfx1250) — matrix operand order repaired and checked by reading, not compiled, not run on hardware
//
#include <hip/hip_runtime.h>
#include <hip/hip_bf16.h>

// ---------------------------------------------------------------------------
// Types for CDNA5 WMMA (wave32): v_wmma_f32_16x16x32_bf16
// ---------------------------------------------------------------------------
typedef __attribute__((ext_vector_type(16))) __bf16 v16bf;
typedef __attribute__((ext_vector_type(8)))  float  v8f;

union FragBF16 {
    v16bf v;
    uint4 q[2];   // 32 bytes
};

#define MROWS 4096          // B*P = 2*2048 rows through the MLP
#define IMG_H 512
#define IMG_W 512
#define WSZ   32            // 32x32 sampling window -> 1024 features

// ---------------------------------------------------------------------------
// Bicubic kernel weight (order=4 branch of the reference)
// ---------------------------------------------------------------------------
__device__ __forceinline__ float cubicw(float s1) {
    float s = s1 + 1e-6f;
    float a = fabsf(s);
    float o2 = ((-0.5f * a + 2.5f) * a - 4.0f) * a + 2.0f;  // 1 <= |s| < 2
    float o1 = (1.5f * a - 2.5f) * a * a + 1.0f;            // |s| < 1
    float oeq = (o1 + 3.0f * o2) * 0.25f;
    float out = (a < 2.0f) ? o2 : 0.0f;
    out = (a == 1.0f) ? oeq : out;
    out = (a < 1.0f) ? o1 : out;
    return out;
}

__device__ __forceinline__ float reflect_coord(float x, float mn, float mx) {
    x = (x > mx) ? (2.0f * mx - x) : x;
    x = (x < mn) ? (2.0f * mn - x) : x;
    return x;
}

// ---------------------------------------------------------------------------
// Stage 1: bicubic grid sample of 32x32 windows -> bf16 activations (4096x1024)
// One thread per (row, window-tap).
// ---------------------------------------------------------------------------
__global__ __launch_bounds__(256)
void funknn_sample_kernel(const float* __restrict__ img,
                          const float* __restrict__ coords,
                          const float* __restrict__ ws1,
                          __hip_bfloat16* __restrict__ h0) {
    int idx = blockIdx.x * 256 + threadIdx.x;    // [0, 4096*1024)
    int k   = idx & (WSZ * WSZ - 1);             // window tap
    int row = idx >> 10;                         // 0..4095
    int b   = row >> 11;                         // batch
    int p   = row & 2047;                        // point

    int j = k & (WSZ - 1);                       // x offset index
    int i = k >> 5;                              // y offset index
    float oj = (float)j - 15.5f;
    float oi = (float)i - 15.5f;

    float scale = ws1[0] * (2.0f / (float)IMG_H);
    float gx = coords[(b * 2048 + p) * 2 + 0] + scale * oj;
    float gy = coords[(b * 2048 + p) * 2 + 1] + scale * oi;

    float ix = (1.0f + gx) * ((float)IMG_W * 0.5f) - 0.5f;
    float iy = (1.0f + gy) * ((float)IMG_H * 0.5f) - 0.5f;
    float ixb = floorf(ix);
    float iyb = floorf(iy);

    float wx[4], wy[4];
    int   xi[4], yi[4];
#pragma unroll
    for (int t = 0; t < 4; ++t) {
        float px = ixb + (float)(t - 1);
        float py = iyb + (float)(t - 1);
        wx[t] = cubicw(ix - px);
        wy[t] = cubicw(iy - py);
        float rx = reflect_coord(px, -0.5f, (float)IMG_W - 0.5f);
        float ry = reflect_coord(py, -0.5f, (float)IMG_H - 0.5f);
        int xt = (int)rx;  xt = xt < 0 ? 0 : (xt > IMG_W - 1 ? IMG_W - 1 : xt);
        int yt = (int)ry;  yt = yt < 0 ? 0 : (yt > IMG_H - 1 ? IMG_H - 1 : yt);
        xi[t] = xt; yi[t] = yt;
    }

    const float* im = img + (size_t)b * IMG_H * IMG_W;
    float acc = 0.0f;
#pragma unroll
    for (int a = 0; a < 4; ++a) {
#pragma unroll
        for (int c = 0; c < 4; ++c) {
            acc = fmaf(wx[a] * wy[c], im[yi[c] * IMG_W + xi[a]], acc);
        }
    }
    h0[(size_t)row * (WSZ * WSZ) + k] = __float2bfloat16(acc);
}

// ---------------------------------------------------------------------------
// Weight convert+transpose: W (K x N, f32 row-major) -> Wt (N x K, bf16)
// so B-matrix WMMA fragments become contiguous 128-bit loads along K.
// ---------------------------------------------------------------------------
__global__ __launch_bounds__(256)
void funknn_wconv_kernel(const float* __restrict__ W,
                         __hip_bfloat16* __restrict__ Wt,
                         int K, int N) {
    int idx = blockIdx.x * 256 + threadIdx.x;
    if (idx >= K * N) return;
    int k = idx / N;
    int n = idx - k * N;
    Wt[(size_t)n * K + k] = __float2bfloat16(W[idx]);
}

// ---------------------------------------------------------------------------
// Stage 2: WMMA GEMM  O = relu(A @ W + bias),  A: M x K bf16, Wt: N x K bf16.
// Block = 256 threads = 8 waves stacked in M (BM = 128, BN = NT*16).
// All 8 waves share the block's B panel, so the panel is staged in LDS once
// per K-step (8x less L2 B-traffic) in a double-buffered pipeline with one
// barrier per step: iteration i reads buf i%2 and writes buf (i+1)%2, with
// the next panel's global load issued BEFORE the WMMAs so its latency hides
// behind the matrix pipe. K is a template constant and the K-loop is fully
// unrolled: fresh SSA fragments (no mov rotation) and compile-time immediate
// load offsets (no per-step address math). All NT B-fragments are DS-loaded
// before the first WMMA of a step so the 2*NT ds_load_b128 issue as one
// clause with staggered dscnt waits instead of a full wait per tile.
// For BN==64 every thread is a B-loader, so the predicate folds away and no
// EXEC save/restore is emitted around the staging code.
// LDS layout: column-major, 80B stride/col -> 16B aligned for b128 DS reads
// and (80/4=20) conflict-free across lanes 0..15.
// Fragment layouts follow CDNA5 ISA 7.12.2 (wave32):
//   A 16x32 bf16 : lane l=lane&15 -> row; halves[0..7]=K(kb+8*hi..),
//                  halves[8..15]=K(kb+16+8*hi..)
//   B 32x16 bf16 : lane l -> col;  halves[0..7]=K(kb+8*hi..), halves[8..15]=K(kb+16+8*hi..), same as A
//   D 16x16 f32  : elem v -> row v+8*hi, lane l -> col
// ---------------------------------------------------------------------------
template <int NT, int K>
__global__ __launch_bounds__(256)
void funknn_gemm_kernel(const __hip_bfloat16* __restrict__ A,
                        const __hip_bfloat16* __restrict__ Wt,
                        const float* __restrict__ bias,
                        __hip_bfloat16* __restrict__ O,
                        int N) {
    constexpr int  BN       = NT * 16;    // block column-tile
    constexpr int  BSTRIDE  = 80;         // bytes per staged column (64 + 16 pad)
    constexpr bool ALL_LOAD = (BN == 64); // 256 threads == BN*4 loaders

    __shared__ __align__(16) unsigned char smemB[2][BN][BSTRIDE];

    const int tid  = threadIdx.x;
    const int wave = tid >> 5;
    const int lane = tid & 31;
    const int l    = lane & 15;
    const int hi   = lane >> 4;

    const int m0 = blockIdx.x * 128 + wave * 16;     // row base of this wave
    const int n0 = blockIdx.y * BN;                  // col base of this block

    const __hip_bfloat16* arow = A + (size_t)(m0 + l) * K + 8 * hi;

    // Cooperative B loader: thread tid stages 16B of column (tid>>2), chunk
    // (tid&3). Predicate is wave-uniform; for BN==64 it is constant-true.
    const int  lcol   = tid >> 2;
    const int  lq     = tid & 3;
    const bool loader = ALL_LOAD || (lcol < BN);
    const __hip_bfloat16* bsrc =
        Wt + (size_t)(n0 + (loader ? lcol : 0)) * K + lq * 8;

    // Prologue: stage B(kb=0) into buffer 0.
    if (loader) {
        uint4 g0 = *reinterpret_cast<const uint4*>(bsrc);
        *reinterpret_cast<uint4*>(&smemB[0][lcol][lq * 16]) = g0;
    }
    __syncthreads();

    v8f acc[NT];
#pragma unroll
    for (int t = 0; t < NT; ++t) acc[t] = (v8f){0.f, 0.f, 0.f, 0.f, 0.f, 0.f, 0.f, 0.f};

#pragma unroll
    for (int kb = 0; kb < K; kb += 32) {
        const int  buf      = (kb >> 5) & 1;
        const bool haveNext = (kb + 32 < K);

        // (1) Issue next panel's global load before any compute.
        uint4 g = {0u, 0u, 0u, 0u};
        if (haveNext && loader)
            g = *reinterpret_cast<const uint4*>(bsrc + kb + 32);

        // (2) A fragment: direct global, compile-time immediate offsets.
        FragBF16 a;
        a.q[0] = *reinterpret_cast<const uint4*>(arow + kb);
        a.q[1] = *reinterpret_cast<const uint4*>(arow + kb + 16);

        // (3) All B fragments from LDS first (one DS clause), then WMMAs.
        FragBF16 b[NT];
#pragma unroll
        for (int t = 0; t < NT; ++t) {
            const unsigned char* bp = &smemB[buf][t * 16 + l][hi * 16];
            b[t].q[0] = *reinterpret_cast<const uint4*>(bp);
            b[t].q[1] = *reinterpret_cast<const uint4*>(bp + 32);
        }
#pragma unroll
        for (int t = 0; t < NT; ++t) {
            acc[t] = __builtin_amdgcn_wmma_f32_16x16x32_bf16(
                /*neg_a=*/false, a.v, /*neg_b=*/false, b[t].v,
                /*c_mod=*/(short)0, acc[t],
                /*reuse_a=*/false, /*reuse_b=*/false);
        }

        // (4) Commit next panel; single barrier per K-step covers both
        //     write->read (next iter) and read->write (this buf) hazards.
        if (haveNext) {
            if (loader)
                *reinterpret_cast<uint4*>(&smemB[buf ^ 1][lcol][lq * 16]) = g;
            __syncthreads();
        }
    }

    // Epilogue: bias + relu, store bf16 for the next layer.
#pragma unroll
    for (int t = 0; t < NT; ++t) {
        const int col = n0 + t * 16 + l;
        const float bv = bias[col];
#pragma unroll
        for (int v = 0; v < 8; ++v) {
            float x = acc[t][v] + bv;
            x = fmaxf(x, 0.0f);
            O[(size_t)(m0 + v + 8 * hi) * N + col] = __float2bfloat16(x);
        }
    }
}

// ---------------------------------------------------------------------------
// Tail: layers 21..23 (16->16 relu, 16->16 relu, 16->1). K=16 < WMMA K, and
// total FLOPs are negligible -> one thread per output row, plain VALU.
// ---------------------------------------------------------------------------
__global__ __launch_bounds__(256)
void funknn_tail_kernel(const __hip_bfloat16* __restrict__ act,  // 4096 x 16
                        const float* __restrict__ W21, const float* __restrict__ b21,
                        const float* __restrict__ W22, const float* __restrict__ b22,
                        const float* __restrict__ W23, const float* __restrict__ b23,
                        float* __restrict__ out) {
    int row = blockIdx.x * 256 + threadIdx.x;
    if (row >= MROWS) return;

    float h[16], g[16];
#pragma unroll
    for (int i = 0; i < 16; ++i) h[i] = __bfloat162float(act[(size_t)row * 16 + i]);

#pragma unroll
    for (int n = 0; n < 16; ++n) g[n] = b21[n];
    for (int k = 0; k < 16; ++k) {
        float hk = h[k];
#pragma unroll
        for (int n = 0; n < 16; ++n) g[n] = fmaf(hk, W21[k * 16 + n], g[n]);
    }
#pragma unroll
    for (int n = 0; n < 16; ++n) g[n] = fmaxf(g[n], 0.0f);

#pragma unroll
    for (int n = 0; n < 16; ++n) h[n] = b22[n];
    for (int k = 0; k < 16; ++k) {
        float gk = g[k];
#pragma unroll
        for (int n = 0; n < 16; ++n) h[n] = fmaf(gk, W22[k * 16 + n], h[n]);
    }

    float o = b23[0];
#pragma unroll
    for (int k = 0; k < 16; ++k) o = fmaf(fmaxf(h[k], 0.0f), W23[k], o);
    out[row] = o;
}

// ---------------------------------------------------------------------------
// Host side
// ---------------------------------------------------------------------------
extern "C" void kernel_launch(void* const* d_in, const int* in_sizes, int n_in,
                              void* d_out, int out_size, void* d_ws, size_t ws_size,
                              hipStream_t stream) {
    (void)in_sizes; (void)n_in; (void)out_size; (void)ws_size;

    // Input order: image, coords, ws1, weights[0..23], biases[0..23]
    const float* d_img    = (const float*)d_in[0];
    const float* d_coords = (const float*)d_in[1];
    const float* d_ws1    = (const float*)d_in[2];
    const float* const* dW = (const float* const*)(d_in + 3);
    const float* const* dB = (const float* const*)(d_in + 3 + 24);

    static const int dims[25] = {1024,
        1024, 1024, 1024, 1024,
        512, 512, 512, 512,
        256, 256, 256,
        128, 128, 128,
        64, 64, 64,
        32, 32, 32,
        16, 16, 16,
        1};

    // Workspace layout (bytes): actA | actB | transposed bf16 weights (layers 0..20)
    const size_t ACT_BYTES = (size_t)MROWS * 1024 * sizeof(__hip_bfloat16); // 8 MB
    char* wsb = (char*)d_ws;
    __hip_bfloat16* actA = (__hip_bfloat16*)(wsb);
    __hip_bfloat16* actB = (__hip_bfloat16*)(wsb + ACT_BYTES);
    __hip_bfloat16* wtb  = (__hip_bfloat16*)(wsb + 2 * ACT_BYTES);

    // 1) Convert + transpose weights for the 21 WMMA layers.
    {
        size_t off = 0;
        for (int i = 0; i <= 20; ++i) {
            int K = dims[i], N = dims[i + 1];
            int total = K * N;
            int blocks = (total + 255) / 256;
            funknn_wconv_kernel<<<blocks, 256, 0, stream>>>(dW[i], wtb + off, K, N);
            off += (size_t)total;
        }
    }

    // 2) Bicubic sampling -> h0 (4096 x 1024 bf16).
    {
        int total = MROWS * WSZ * WSZ;
        funknn_sample_kernel<<<total / 256, 256, 0, stream>>>(d_img, d_coords, d_ws1, actA);
    }

    // 3) WMMA GEMM chain for layers 0..20 (ping-pong activation buffers).
    {
        __hip_bfloat16* in   = actA;
        __hip_bfloat16* outb = actB;
        size_t off = 0;
        for (int i = 0; i <= 20; ++i) {
            int K = dims[i], N = dims[i + 1];
            int NT = (N >= 64) ? 4 : (N / 16);     // 4, 2, or 1 col-tiles per wave
            dim3 grid(MROWS / 128, N / (NT * 16));
            dim3 block(256);

#define FUNKNN_GEMM_CASE(nt, kk)                                                     \
            if (NT == (nt) && K == (kk)) {                                           \
                funknn_gemm_kernel<nt, kk><<<grid, block, 0, stream>>>(              \
                    in, wtb + off, dB[i], outb, N);                                  \
            }
            FUNKNN_GEMM_CASE(4, 1024)
            FUNKNN_GEMM_CASE(4, 512)
            FUNKNN_GEMM_CASE(4, 256)
            FUNKNN_GEMM_CASE(4, 128)
            FUNKNN_GEMM_CASE(4, 64)
            FUNKNN_GEMM_CASE(2, 64)
            FUNKNN_GEMM_CASE(2, 32)
            FUNKNN_GEMM_CASE(1, 32)
#undef FUNKNN_GEMM_CASE

            off += (size_t)K * N;
            __hip_bfloat16* tmp = in; in = outb; outb = tmp;
        }
        // 4) Scalar tail: layers 21..23 -> f32 output (4096 x 1).
        funknn_tail_kernel<<<MROWS / 256, 256, 0, stream>>>(
            in, dW[21], dB[21], dW[22], dB[22], dW[23], dB[23], (float*)d_out);
    }
}
